// BertLayer_42013370089869
// MI455X (gfx1250) — compile-verified, matrix operand order checked by reading, not run on hardware
//
#include <hip/hip_runtime.h>
#include <stdint.h>

// ---------------------------------------------------------------------------
// BERT layer forward for MI455X (gfx1250), bf16 WMMA everywhere.
// B=16 S=512 D=1024 H=16 F=4096 DH=64.  ~240 GFLOP, compute-bound =>
// fp32 -> bf16 once, all GEMMs via V_WMMA_F32_16X16X32_BF16 (f32 accum).
// GEMM: double-buffered LDS fed by GLOBAL_LOAD_ASYNC_TO_LDS_B128 (ASYNCcnt),
// one workgroup barrier per 32-deep k-step; epilogue modes are compile-time
// so stores are branch-free. Workspace requirement: ~302 MB.
// ---------------------------------------------------------------------------

typedef __bf16 bf16;
typedef __attribute__((ext_vector_type(16))) __bf16 v16bf;
typedef __attribute__((ext_vector_type(8)))  float  v8f;

#define LDP 40  // LDS row pitch in bf16 elems (80B = 5 x 16B) -> bank spread

// CDNA5 async copy: 16B global -> LDS, tracked by ASYNCcnt, no VGPR bounce.
// LDS address is the low 32 bits of the flat pointer.
__device__ __forceinline__ void async_b128(unsigned lds_addr, const bf16* gaddr) {
  asm volatile("global_load_async_to_lds_b128 %0, %1, off"
               :: "v"(lds_addr), "v"(gaddr) : "memory");
}
__device__ __forceinline__ void wait_async0() {
  asm volatile("s_wait_asynccnt 0x0" ::: "memory");
}
__device__ __forceinline__ unsigned lds_addr_of(const bf16* p) {
  return (unsigned)(size_t)p;
}

// Build a 16x32 bf16 WMMA fragment from an LDS tile row.
// Per ISA 7.12.2: lane half h = lane>>4 picks K chunks {h*8..} and {16+h*8..},
// both 16B contiguous in our LDS layout.
__device__ __forceinline__ v16bf lds_frag(const bf16* base, int row, int half) {
  const bf16* p = base + (size_t)row * LDP;
  v16bf f;
  __builtin_memcpy((char*)&f,      p + half * 8,      16);
  __builtin_memcpy((char*)&f + 16, p + 16 + half * 8, 16);
  return f;
}

// ---------------------------------------------------------------------------
// Generic bf16 GEMM:  C[M,N] = scale * (A[M,K] * Bt[N,K]^T) + bias[n]
// Bt is row-major [N,K] (always A @ B^T).  Batched over blockIdx.z with
// (b,h) decode: bh = batch0+z, b=bh/Hdiv, h=bh%Hdiv.
// EPI: 0=none, 1=exact-erf GELU.  HASF/HASB/HASBIAS: compile-time epilogue.
// BM must be 128 (2 staging chunks/thread); BN in {64,128}.
// ---------------------------------------------------------------------------
template<int BM, int BN, int WTM, int WTN, int WAVES_N, int EPI,
         bool HASF, bool HASB, bool HASBIAS>
__global__ __launch_bounds__(256) void gemm_bf16_wmma(
    const bf16* __restrict__ A, const bf16* __restrict__ Bt,
    float* __restrict__ Cf, bf16* __restrict__ Cb,
    const float* __restrict__ bias,
    int K, int lda, int ldb, int ldc, float scale,
    int batch0, int Hdiv,
    long long sAb, long long sAh, long long sAz,
    long long sBb, long long sBh, long long sBz,
    long long sCb, long long sCh, long long sCz,
    long long sBiasB)
{
  constexpr int WAVES_M = 8 / WAVES_N;
  static_assert(WAVES_M * WTM * 16 == BM, "tile M mismatch");
  static_assert(WAVES_N * WTN * 16 == BN, "tile N mismatch");
  static_assert(BM == 128, "staging assumes BM==128");
  constexpr int BCH = (BN * 4) / 256;  // B-tile 16B chunks per thread (1 or 2)

  __shared__ __align__(16) bf16 sA[2][BM * LDP];
  __shared__ __align__(16) bf16 sB[2][BN * LDP];

  const int tid  = threadIdx.x;
  const int lane = tid & 31;
  const int wave = tid >> 5;
  const int wm   = wave / WAVES_N;
  const int wn   = wave % WAVES_N;
  const int half = lane >> 4;
  const int lrow = lane & 15;

  const int z  = blockIdx.z;
  const int bh = batch0 + z;
  const int b  = bh / Hdiv;
  const int h  = bh % Hdiv;
  const long long aoff = (long long)b * sAb + (long long)h * sAh + (long long)z * sAz;
  const long long boff = (long long)b * sBb + (long long)h * sBh + (long long)z * sBz;
  const long long coff = (long long)b * sCb + (long long)h * sCh + (long long)z * sCz;

  const int row0 = blockIdx.y * BM;
  const int col0 = blockIdx.x * BN;

  // ---- per-thread staging addresses (computed once; k-loop just adds 32) ----
  const int r0 = tid >> 2, c0 = tid & 3;   // chunk (row, 8-elem col) in tile
  const bf16* gA0 = A  + aoff + (long long)(row0 + r0) * lda + c0 * 8;
  const bf16* gA1 = gA0 + (long long)64 * lda;                 // rows 64..127
  const bf16* gB0 = Bt + boff + (long long)(col0 + r0) * ldb + c0 * 8;
  const bf16* gB1 = gB0 + (long long)64 * ldb;                 // only if BCH==2

  unsigned lA0[2], lA1[2], lB0[2], lB1[2];
  #pragma unroll
  for (int u = 0; u < 2; ++u) {
    lA0[u] = lds_addr_of(&sA[u][r0 * LDP + c0 * 8]);
    lA1[u] = lds_addr_of(&sA[u][(r0 + 64) * LDP + c0 * 8]);
    lB0[u] = lds_addr_of(&sB[u][r0 * LDP + c0 * 8]);
    if (BCH == 2) lB1[u] = lds_addr_of(&sB[u][(r0 + 64) * LDP + c0 * 8]);
  }

  v8f acc[WTM][WTN];
  #pragma unroll
  for (int i = 0; i < WTM; ++i)
    #pragma unroll
    for (int j = 0; j < WTN; ++j)
      #pragma unroll
      for (int e = 0; e < 8; ++e) acc[i][j][e] = 0.0f;

  const int KT = K / 32;

  // Prologue: stage k-tile 0 into buffer 0.
  async_b128(lA0[0], gA0);
  async_b128(lA1[0], gA1);
  async_b128(lB0[0], gB0);
  if constexpr (BCH == 2) async_b128(lB1[0], gB1);
  wait_async0();
  __syncthreads();

  int cur = 0;
  for (int kt = 0; kt < KT; ++kt) {
    const int nxt = cur ^ 1;
    if (kt + 1 < KT) {           // uniform
      gA0 += 32; gA1 += 32; gB0 += 32;
      async_b128(lA0[nxt], gA0);
      async_b128(lA1[nxt], gA1);
      async_b128(lB0[nxt], gB0);
      if constexpr (BCH == 2) { gB1 += 32; async_b128(lB1[nxt], gB1); }
    }

    // Compute on current buffer while next tile streams in.
    v16bf afr[WTM], bfr[WTN];
    #pragma unroll
    for (int mt = 0; mt < WTM; ++mt)
      afr[mt] = lds_frag(&sA[cur][0], wm * WTM * 16 + mt * 16 + lrow, half);
    #pragma unroll
    for (int nt = 0; nt < WTN; ++nt)
      bfr[nt] = lds_frag(&sB[cur][0], wn * WTN * 16 + nt * 16 + lrow, half);

    #pragma unroll
    for (int mt = 0; mt < WTM; ++mt)
      #pragma unroll
      for (int nt = 0; nt < WTN; ++nt)
        acc[mt][nt] = __builtin_amdgcn_wmma_f32_16x16x32_bf16(
            false, afr[mt], false, bfr[nt], (short)0, acc[mt][nt], false, false);

    wait_async0();               // free when count already 0
    __syncthreads();
    cur = nxt;
  }

  // Epilogue. C/D 16x16 layout: VGPR r, lane l: m = r + (l>=16 ? 8 : 0), n = l&15.
  // Branch-free: output/bias selection is compile-time; one 64-bit base per
  // accumulator tile, rows walked by ptr += ldc.
  #pragma unroll
  for (int nt = 0; nt < WTN; ++nt) {
    const long long col = col0 + wn * WTN * 16 + nt * 16 + lrow;
    float bv = 0.0f;
    if constexpr (HASBIAS) bv = bias[(long long)b * sBiasB + col];
    #pragma unroll
    for (int mt = 0; mt < WTM; ++mt) {
      const long long rowBase = row0 + wm * WTM * 16 + mt * 16 + (half ? 8 : 0);
      const long long base = coff + rowBase * ldc + col;
      float* pf = Cf + base;
      bf16*  pb = Cb + base;
      #pragma unroll
      for (int r = 0; r < 8; ++r) {
        float v = acc[mt][nt][r] * scale + bv;
        if constexpr (EPI == 1) v = 0.5f * v * (1.0f + erff(v * 0.70710678118654752f));
        if constexpr (HASF) *pf = v;
        if constexpr (HASB) *pb = (bf16)v;
        pf += ldc;
        pb += ldc;
      }
    }
  }
}

// ---------------------------------------------------------------------------
// Row softmax over 512 columns (scores already have scale+mask applied).
// ---------------------------------------------------------------------------
__global__ __launch_bounds__(256) void softmax512(
    const float* __restrict__ Sc, bf16* __restrict__ P)
{
  __shared__ float red[256];
  long long row = blockIdx.x;
  const float* x = Sc + row * 512;
  bf16* y = P + row * 512;
  int tid = threadIdx.x;
  float v0 = x[tid], v1 = x[tid + 256];
  red[tid] = fmaxf(v0, v1);
  __syncthreads();
  for (int s = 128; s > 0; s >>= 1) { if (tid < s) red[tid] = fmaxf(red[tid], red[tid + s]); __syncthreads(); }
  float m = red[0];
  __syncthreads();
  float e0 = __expf(v0 - m), e1 = __expf(v1 - m);
  red[tid] = e0 + e1;
  __syncthreads();
  for (int s = 128; s > 0; s >>= 1) { if (tid < s) red[tid] += red[tid + s]; __syncthreads(); }
  float inv = 1.0f / red[0];
  y[tid]       = (bf16)(e0 * inv);
  y[tid + 256] = (bf16)(e1 * inv);
}

// ---------------------------------------------------------------------------
// y = LN(C + bias + resid) * g + b over D=1024; fp32 out (+ optional bf16 out).
// ---------------------------------------------------------------------------
__global__ __launch_bounds__(256) void bias_resid_ln1024(
    const float* __restrict__ C, const float* __restrict__ resid,
    const float* __restrict__ bias,
    const float* __restrict__ g, const float* __restrict__ bb,
    float* __restrict__ outF, bf16* __restrict__ outB)
{
  __shared__ float red[256];
  long long row = blockIdx.x;
  int tid = threadIdx.x;
  float v[4];
  #pragma unroll
  for (int i = 0; i < 4; ++i) {
    int idx = tid + i * 256;
    v[i] = C[row * 1024 + idx] + bias[idx] + resid[row * 1024 + idx];
  }
  red[tid] = v[0] + v[1] + v[2] + v[3];
  __syncthreads();
  for (int s = 128; s > 0; s >>= 1) { if (tid < s) red[tid] += red[tid + s]; __syncthreads(); }
  float mean = red[0] * (1.0f / 1024.0f);
  __syncthreads();
  float q = 0.0f;
  #pragma unroll
  for (int i = 0; i < 4; ++i) { float d = v[i] - mean; q += d * d; }
  red[tid] = q;
  __syncthreads();
  for (int s = 128; s > 0; s >>= 1) { if (tid < s) red[tid] += red[tid + s]; __syncthreads(); }
  float rstd = rsqrtf(red[0] * (1.0f / 1024.0f) + 1e-12f);
  #pragma unroll
  for (int i = 0; i < 4; ++i) {
    int idx = tid + i * 256;
    float yv = (v[i] - mean) * rstd * g[idx] + bb[idx];
    outF[row * 1024 + idx] = yv;
    if (outB) outB[row * 1024 + idx] = (bf16)yv;
  }
}

// fp32 -> bf16 elementwise
__global__ __launch_bounds__(256) void cvt_bf16(
    const float* __restrict__ in, bf16* __restrict__ out, long long n)
{
  long long base = (long long)blockIdx.x * 1024 + threadIdx.x;
  #pragma unroll
  for (int i = 0; i < 4; ++i) {
    long long idx = base + i * 256;
    if (idx < n) out[idx] = (bf16)in[idx];
  }
}

// fp32 [K,N] -> bf16 [N,K] (weight transpose so GEMM always sees Bt[N,K])
__global__ __launch_bounds__(256) void cvt_transpose_bf16(
    const float* __restrict__ in, bf16* __restrict__ out, int Kd, int Nd)
{
  long long idx = (long long)blockIdx.x * 256 + threadIdx.x;
  if (idx < (long long)Kd * Nd) {
    int k = (int)(idx / Nd), n = (int)(idx % Nd);
    out[(long long)n * Kd + k] = (bf16)in[idx];
  }
}

// Vb [b*S+s, h*DH+d] (bf16, D=1024) -> Vt [(b*H+h)*DH*S + d*S + s]
// LDS-tiled 64x64 so both global accesses are coalesced.
__global__ __launch_bounds__(256) void transpose_v(
    const bf16* __restrict__ Vb, bf16* __restrict__ Vt)
{
  __shared__ bf16 t[64][72];
  int bh = blockIdx.x;
  int b = bh >> 4, h = bh & 15;
  for (int s0 = 0; s0 < 512; s0 += 64) {
    for (int i = threadIdx.x; i < 64 * 64; i += 256) {
      int s = i >> 6, d = i & 63;                       // read: d contiguous
      t[d][s] = Vb[(long long)(b * 512 + s0 + s) * 1024 + h * 64 + d];
    }
    __syncthreads();
    for (int i = threadIdx.x; i < 64 * 64; i += 256) {
      int d = i >> 6, s = i & 63;                       // write: s contiguous
      Vt[(long long)bh * 32768 + (long long)d * 512 + s0 + s] = t[d][s];
    }
    __syncthreads();
  }
}

// ---------------------------------------------------------------------------
extern "C" void kernel_launch(void* const* d_in, const int* in_sizes, int n_in,
                              void* d_out, int out_size, void* d_ws, size_t ws_size,
                              hipStream_t stream) {
  (void)in_sizes; (void)n_in; (void)out_size; (void)ws_size;
  constexpr int Bn = 16, S = 512, D = 1024, H = 16, F = 4096, DH = 64;
  constexpr int M = Bn * S;           // 8192 rows
  constexpr int CH = 16;              // (b,h) pairs per attention chunk

  const float* x    = (const float*)d_in[0];
  const float* mask = (const float*)d_in[1];
  const float* Wq   = (const float*)d_in[2];  const float* bq  = (const float*)d_in[3];
  const float* Wk   = (const float*)d_in[4];  const float* bk  = (const float*)d_in[5];
  const float* Wv   = (const float*)d_in[6];  const float* bv  = (const float*)d_in[7];
  const float* Wo   = (const float*)d_in[8];  const float* bo  = (const float*)d_in[9];
  const float* ln1g = (const float*)d_in[10]; const float* ln1b = (const float*)d_in[11];
  const float* Wi   = (const float*)d_in[12]; const float* bi  = (const float*)d_in[13];
  const float* Wo2  = (const float*)d_in[14]; const float* bo2 = (const float*)d_in[15];
  const float* ln2g = (const float*)d_in[16]; const float* ln2b = (const float*)d_in[17];
  float* out = (float*)d_out;

  char* w = (char*)d_ws;
  size_t off = 0;
  auto take = [&](size_t bytes) -> char* {
    char* p = w + off;
    off = (off + bytes + 255) & ~(size_t)255;
    return p;
  };
  bf16*  Xb      = (bf16*)take((size_t)M * D * 2);
  bf16*  Wqt     = (bf16*)take((size_t)D * D * 2);
  bf16*  Wkt     = (bf16*)take((size_t)D * D * 2);
  bf16*  Wvt     = (bf16*)take((size_t)D * D * 2);
  bf16*  Wot     = (bf16*)take((size_t)D * D * 2);
  bf16*  Wit     = (bf16*)take((size_t)D * F * 2);
  bf16*  Wo2t    = (bf16*)take((size_t)D * F * 2);
  bf16*  Qb      = (bf16*)take((size_t)M * D * 2);
  bf16*  Kb      = (bf16*)take((size_t)M * D * 2);
  bf16*  Vbb     = (bf16*)take((size_t)M * D * 2);
  bf16*  Vt      = (bf16*)take((size_t)M * D * 2);
  bf16*  Ctxb    = (bf16*)take((size_t)M * D * 2);
  float* Tmp     = (float*)take((size_t)M * D * 4);   // reused for both dense outs
  float* AttnOut = (float*)take((size_t)M * D * 4);
  bf16*  AttnOutB= (bf16*)take((size_t)M * D * 2);
  bf16*  InterB  = (bf16*)take((size_t)M * F * 2);
  float* ScoresC = (float*)take((size_t)CH * S * S * 4);
  bf16*  ProbsC  = (bf16*)take((size_t)CH * S * S * 2);

  // --- precision conversion + weight transposition (once per call) ---
  cvt_bf16<<<(M * D + 1023) / 1024, 256, 0, stream>>>(x, Xb, (long long)M * D);
  cvt_transpose_bf16<<<(D * D) / 256, 256, 0, stream>>>(Wq, Wqt, D, D);
  cvt_transpose_bf16<<<(D * D) / 256, 256, 0, stream>>>(Wk, Wkt, D, D);
  cvt_transpose_bf16<<<(D * D) / 256, 256, 0, stream>>>(Wv, Wvt, D, D);
  cvt_transpose_bf16<<<(D * D) / 256, 256, 0, stream>>>(Wo, Wot, D, D);
  cvt_transpose_bf16<<<(D * F) / 256, 256, 0, stream>>>(Wi, Wit, D, F);
  cvt_transpose_bf16<<<(D * F) / 256, 256, 0, stream>>>(Wo2, Wo2t, F, D);

  // --- QKV projections (bf16 out, bias fused) ---
  dim3 blk(256);
  {
    dim3 grid(D / 128, M / 128, 1);
    gemm_bf16_wmma<128, 128, 4, 2, 4, 0, false, true, true><<<grid, blk, 0, stream>>>(
        Xb, Wqt, nullptr, Qb, bq, D, D, D, D, 1.0f, 0, 1,
        0, 0, 0, 0, 0, 0, 0, 0, 0, 0);
    gemm_bf16_wmma<128, 128, 4, 2, 4, 0, false, true, true><<<grid, blk, 0, stream>>>(
        Xb, Wkt, nullptr, Kb, bk, D, D, D, D, 1.0f, 0, 1,
        0, 0, 0, 0, 0, 0, 0, 0, 0, 0);
    gemm_bf16_wmma<128, 128, 4, 2, 4, 0, false, true, true><<<grid, blk, 0, stream>>>(
        Xb, Wvt, nullptr, Vbb, bv, D, D, D, D, 1.0f, 0, 1,
        0, 0, 0, 0, 0, 0, 0, 0, 0, 0);
  }
  transpose_v<<<Bn * H, 256, 0, stream>>>(Vbb, Vt);

  // --- attention, CH (b,h)-pairs per chunk (stream-serialized, reuses ws) ---
  for (int c = 0; c < (Bn * H) / CH; ++c) {
    int batch0 = c * CH;
    // scores = Q K^T / sqrt(DH) + mask[b, key]
    dim3 gs(S / 128, S / 128, CH);
    gemm_bf16_wmma<128, 128, 4, 2, 4, 0, true, false, true><<<gs, blk, 0, stream>>>(
        Qb, Kb, ScoresC, nullptr, mask, DH, D, D, S, 0.125f, batch0, H,
        (long long)S * D, DH, 0,
        (long long)S * D, DH, 0,
        0, 0, (long long)S * S,
        S);
    softmax512<<<CH * S, 256, 0, stream>>>(ScoresC, ProbsC);
    // ctx = P V  (N = 64 tile config)
    dim3 gc(DH / 64, S / 128, CH);
    gemm_bf16_wmma<128, 64, 2, 2, 2, 0, false, true, false><<<gc, blk, 0, stream>>>(
        ProbsC, Vt, nullptr, Ctxb, nullptr, S, S, S, D, 1.0f, batch0, H,
        0, 0, (long long)S * S,
        (long long)H * DH * S, (long long)DH * S, 0,
        (long long)S * D, DH, 0,
        0);
  }

  // --- attention output projection + residual LN ---
  {
    dim3 grid(D / 128, M / 128, 1);
    gemm_bf16_wmma<128, 128, 4, 2, 4, 0, true, false, false><<<grid, blk, 0, stream>>>(
        Ctxb, Wot, Tmp, nullptr, nullptr, D, D, D, D, 1.0f, 0, 1,
        0, 0, 0, 0, 0, 0, 0, 0, 0, 0);
  }
  bias_resid_ln1024<<<M, 256, 0, stream>>>(Tmp, x, bo, ln1g, ln1b, AttnOut, AttnOutB);

  // --- FFN: intermediate (GELU fused, bf16 out) ---
  {
    dim3 grid(F / 128, M / 128, 1);
    gemm_bf16_wmma<128, 128, 4, 2, 4, 1, false, true, true><<<grid, blk, 0, stream>>>(
        AttnOutB, Wit, nullptr, InterB, bi, D, D, D, F, 1.0f, 0, 1,
        0, 0, 0, 0, 0, 0, 0, 0, 0, 0);
  }
  // --- FFN: output projection + residual LN -> d_out ---
  {
    dim3 grid(D / 128, M / 128, 1);
    gemm_bf16_wmma<128, 128, 4, 2, 4, 0, true, false, false><<<grid, blk, 0, stream>>>(
        InterB, Wo2t, Tmp, nullptr, nullptr, F, F, F, D, 1.0f, 0, 1,
        0, 0, 0, 0, 0, 0, 0, 0, 0, 0);
  }
  bias_resid_ln1024<<<M, 256, 0, stream>>>(Tmp, AttnOut, bo2, ln2g, ln2b, out, nullptr);
}
